// TorchANI_Local_88441966559570
// MI455X (gfx1250) — matrix operand order repaired and checked by reading, not compiled, not run on hardware
//
#include <hip/hip_runtime.h>
#include <hip/hip_bf16.h>
#include <math.h>

// ---------- types for WMMA (CDNA5 / gfx1250, wave32) ----------
typedef __bf16 bf16_t;
typedef bf16_t v16bf __attribute__((ext_vector_type(16)));
typedef float  v8f   __attribute__((ext_vector_type(8)));

union FragBF {
    v16bf          v;
    unsigned short s[16];
    uint4          q[2];
};

__device__ __forceinline__ unsigned short f2bf(float f) {
    unsigned u = __float_as_uint(f);
    u += 0x7FFFu + ((u >> 16) & 1u);   // round-to-nearest-even
    return (unsigned short)(u >> 16);
}
__device__ __forceinline__ float bf2lo(unsigned u) {   // low 16 bits -> float
    return __uint_as_float(u << 16);
}
__device__ __forceinline__ float bf2hi(unsigned u) {   // high 16 bits -> float
    return __uint_as_float(u & 0xFFFF0000u);
}

// ---------------- constants ----------------
#define NUM_SPECIES 7
#define BATCH 64
#define ATOMS 32
#define NATOMS 2048          // BATCH*ATOMS
#define DPAD 1024            // AEV padded K (valid = 1008)
#define DRAD 112             // 7*16
#define DANG 896             // 28*32

__device__ __forceinline__ float celu01(float x) {
    return x > 0.f ? x : 0.1f * (expf(10.f * x) - 1.f);
}

// =====================================================================
// Kernel 1: AEV per atom (block = one (b,i); 128 threads = 4 waves)
// =====================================================================
__global__ void aev_kernel(const int* __restrict__ species,
                           const float* __restrict__ coords,
                           unsigned short* __restrict__ aev,
                           float* __restrict__ accum) {
    __shared__ float dd[ATOMS], fcr[ATOMS], fca[ATOMS];
    __shared__ float ux[ATOMS], uy[ATOMS], uz[ATOMS];
    __shared__ int   sp[ATOMS];
    __shared__ float rad[DRAD];
    __shared__ float ang[DANG];

    const int atom = blockIdx.x;
    const int b = atom >> 5;
    const int i = atom & 31;
    const int tid = threadIdx.x;

    if (tid == 0) accum[atom] = 0.f;
    for (int t = tid; t < DRAD; t += 128) rad[t] = 0.f;
    for (int t = tid; t < DANG; t += 128) ang[t] = 0.f;

    const float xi = coords[(b * ATOMS + i) * 3 + 0];
    const float yi = coords[(b * ATOMS + i) * 3 + 1];
    const float zi = coords[(b * ATOMS + i) * 3 + 2];

    if (tid < ATOMS) {
        const int j = tid;
        const float x = coords[(b * ATOMS + j) * 3 + 0] - xi;
        const float y = coords[(b * ATOMS + j) * 3 + 1] - yi;
        const float z = coords[(b * ATOMS + j) * 3 + 2] - zi;
        const float d = sqrtf(x * x + y * y + z * z + 1e-12f);
        const float inv = 1.f / d;
        dd[j] = d;
        ux[j] = x * inv; uy[j] = y * inv; uz[j] = z * inv;
        float fr = (d < 5.1f) ? (0.5f * cosf(3.14159265358979f * d / 5.1f) + 0.5f) : 0.f;
        float fa = (d < 3.5f) ? (0.5f * cosf(3.14159265358979f * d / 3.5f) + 0.5f) : 0.f;
        if (j == i) { fr = 0.f; fa = 0.f; }
        fcr[j] = fr; fca[j] = fa;
        sp[j] = species[b * ATOMS + j];
    }
    __syncthreads();

    // ---- radial part: threads 0..31 each handle one neighbor j ----
    if (tid < ATOMS) {
        const int j = tid;
        const float fr = fcr[j];
        if (fr > 0.f) {
            const float d = dd[j];
            const int   s = sp[j];
#pragma unroll
            for (int t = 0; t < 16; ++t) {
                const float dv = d - (0.8f + 0.26875f * (float)t);
                atomicAdd(&rad[s * 16 + t], 0.25f * expf(-19.7f * dv * dv) * fr);
            }
        }
    }

    // ---- angular part: 496 unordered pairs (j<k), x2 symmetry ----
    for (int p = tid; p < 496; p += 128) {
        int j = 0, rem = p;
        while (rem >= 31 - j) { rem -= 31 - j; ++j; }
        const int k = j + 1 + rem;
        const float w2 = fca[j] * fca[k];
        if (w2 > 0.f) {
            float ct = ux[j] * ux[k] + uy[j] * uy[k] + uz[j] * uz[k];
            ct = fminf(1.f, fmaxf(-1.f, ct));
            const float th = acosf(0.95f * ct);
            const float avg = 0.5f * (dd[j] + dd[k]);
            float f2v[8];
#pragma unroll
            for (int t = 0; t < 8; ++t) {
                const float dv = avg - (0.8f + 0.3375f * (float)t);
                f2v[t] = expf(-12.5f * dv * dv);
            }
            const int sa = sp[j] < sp[k] ? sp[j] : sp[k];
            const int sb = sp[j] < sp[k] ? sp[k] : sp[j];
            const int pid = sa * 7 - ((sa * (sa - 1)) >> 1) + (sb - sa);
            const float coef0 = 2.f * w2;
#pragma unroll
            for (int z = 0; z < 4; ++z) {
                const float fz = powf(0.5f * (1.f + cosf(th - 0.78539816339f * ((float)z + 0.5f))), 14.1f);
                const float cz = coef0 * fz;
                const int base = pid * 32 + z * 8;
#pragma unroll
                for (int t = 0; t < 8; ++t)
                    atomicAdd(&ang[base + t], cz * f2v[t]);
            }
        }
    }
    __syncthreads();

    // ---- emit bf16 AEV row, padded to 1024 ----
    unsigned short* outp = aev + (size_t)atom * DPAD;
    for (int t = tid; t < DPAD; t += 128) {
        float v = (t < DRAD) ? rad[t] : ((t < DRAD + DANG) ? ang[t - DRAD] : 0.f);
        outp[t] = f2bf(v);
    }
}

// =====================================================================
// Kernel 2: weight prepack -> WMMA B-fragment layout, bf16, K zero-pad.
// Wt layout: [S][KB][NB][lane(32)][16 halves]  (one 32x16 KxN fragment
// = 512 halves = 1KB, contiguous; each lane's 16 halves contiguous).
// Fragment mapping (per ISA 7.12.2, 16-bit B 32x16):
//   lane l: N = l%16 ; half t: K = kb*32 + (t<8 ? 8*(l>=16) + t : 16 + 8*(l>=16) + t-8)
// =====================================================================
__global__ void prepack_kernel(const float* __restrict__ W,   // [S][Kvalid][N]
                               unsigned short* __restrict__ Wt,
                               int Kvalid, int KB, int NB, int N) {
    const int total = NUM_SPECIES * KB * NB * 32;
    const int idx = blockIdx.x * 256 + threadIdx.x;
    if (idx >= total) return;
    const int lane = idx & 31;
    int rest = idx >> 5;
    const int nb = rest % NB;  rest /= NB;
    const int kb = rest % KB;
    const int s  = rest / KB;
    const int hi  = lane >> 4;
    const int l15 = lane & 15;
    const int n = nb * 16 + l15;
    const float* Wp = W + (size_t)s * Kvalid * N;

    FragBF f;
#pragma unroll
    for (int t = 0; t < 16; ++t) {
        const int k = kb * 32 + (t < 8 ? hi * 8 + t : 16 + hi * 8 + (t - 8));
        f.s[t] = (k < Kvalid) ? f2bf(Wp[(size_t)k * N + n]) : (unsigned short)0;
    }
    unsigned short* o = Wt + ((((size_t)s * KB + kb) * NB + nb) * 32 + lane) * 16;
    *(uint4*)(o)     = f.q[0];
    *(uint4*)(o + 8) = f.q[1];
}

// =====================================================================
// Kernel 3: fused GEMM + bias + CELU via v_wmma_f32_16x16x32_bf16.
// Block = 256 threads (8 waves). Block tile 128(M) x 64(N); each wave
// owns a 32x32 tile = 2x2 WMMA fragments. grid = (16*NT, 7 species).
// Inner loop: 8x global_load_b128 + 4x WMMA, no branches, no converts.
// =====================================================================
__device__ __forceinline__ void store_tile(const v8f& c, int rb, int cb,
                                           int l15, int hi, int N,
                                           const float* __restrict__ bp,
                                           unsigned short* __restrict__ Hs,
                                           int ldo) {
    const int col = cb + l15;
    if (col >= N) return;
    const float bv = bp[col];
#pragma unroll
    for (int v = 0; v < 8; ++v) {
        const int row = rb + v + hi * 8;   // C layout: vgpr v -> M = v (+8 for hi lanes)
        float x = c[v] + bv;
        x = celu01(x);
        Hs[(size_t)row * ldo + col] = f2bf(x);
    }
}

__global__ void __launch_bounds__(256)
gemm_celu_kernel(const unsigned short* __restrict__ A, int lda, long long aStrideS,
                 const unsigned short* __restrict__ Wt,  // prepacked fragments
                 const float* __restrict__ bias,
                 unsigned short* __restrict__ H, int ldo,
                 int KB, int NB, int N) {
    const int lane = threadIdx.x & 31;
    const int wid  = threadIdx.x >> 5;
    const int s    = blockIdx.y;
    const int mtile = blockIdx.x & 15;      // 2048/128 = 16 M tiles
    const int ntile = blockIdx.x >> 4;
    const int hi  = lane >> 4;
    const int l15 = lane & 15;

    const int rowBase = mtile * 128 + (wid >> 1) * 32;
    const int colBase = ntile * 64 + (wid & 1) * 32;

    const unsigned short* As = A + (size_t)s * (size_t)aStrideS
                                 + (size_t)(rowBase + l15) * lda + hi * 8;
    // two B fragment columns for this wave (clamped; garbage masked at store)
    int nb0 = colBase >> 4;
    int nb1 = nb0 + 1;
    if (nb0 > NB - 1) nb0 = NB - 1;
    if (nb1 > NB - 1) nb1 = NB - 1;
    const unsigned short* WtS = Wt + (size_t)s * KB * NB * 512 + (size_t)lane * 16;
    const unsigned short* bp0 = WtS + (size_t)nb0 * 512;
    const unsigned short* bp1 = WtS + (size_t)nb1 * 512;
    const size_t bStep = (size_t)NB * 512;   // advance one K-block

    const float* bpv = bias + (size_t)s * (size_t)N;
    unsigned short* Hs = H + (size_t)s * (size_t)NATOMS * (size_t)ldo;

    v8f c00 = {}, c01 = {}, c10 = {}, c11 = {};

    for (int kb = 0; kb < KB; ++kb) {
        FragBF a0, a1, b0, b1;
        const unsigned short* ap = As + kb * 32;
        a0.q[0] = *(const uint4*)(ap);
        a0.q[1] = *(const uint4*)(ap + 16);
        ap += (size_t)16 * lda;
        a1.q[0] = *(const uint4*)(ap);
        a1.q[1] = *(const uint4*)(ap + 16);

        b0.q[0] = *(const uint4*)(bp0);
        b0.q[1] = *(const uint4*)(bp0 + 8);
        b1.q[0] = *(const uint4*)(bp1);
        b1.q[1] = *(const uint4*)(bp1 + 8);
        bp0 += bStep;
        bp1 += bStep;

        c00 = __builtin_amdgcn_wmma_f32_16x16x32_bf16(false, a0.v, false, b0.v, (short)0, c00, false, false);
        c01 = __builtin_amdgcn_wmma_f32_16x16x32_bf16(false, a0.v, false, b1.v, (short)0, c01, false, false);
        c10 = __builtin_amdgcn_wmma_f32_16x16x32_bf16(false, a1.v, false, b0.v, (short)0, c10, false, false);
        c11 = __builtin_amdgcn_wmma_f32_16x16x32_bf16(false, a1.v, false, b1.v, (short)0, c11, false, false);
    }

    store_tile(c00, rowBase,      colBase,      l15, hi, N, bpv, Hs, ldo);
    store_tile(c01, rowBase,      colBase + 16, l15, hi, N, bpv, Hs, ldo);
    store_tile(c10, rowBase + 16, colBase,      l15, hi, N, bpv, Hs, ldo);
    store_tile(c11, rowBase + 16, colBase + 16, l15, hi, N, bpv, Hs, ldo);
}

// =====================================================================
// Kernel 4: last layer 160 -> 1, only for each atom's own species,
// accumulating the ensemble mean (1/8 per ensemble launch).
// =====================================================================
__global__ void l3_kernel(const unsigned short* __restrict__ h3,
                          const int* __restrict__ species,
                          const float* __restrict__ W3e,   // [S][160]
                          const float* __restrict__ b3e,   // [S]
                          float* __restrict__ accum) {
    const int atom = blockIdx.x * 256 + threadIdx.x;
    if (atom >= NATOMS) return;
    const int s = species[atom];
    const uint4*  hv = (const uint4*)(h3 + ((size_t)s * NATOMS + atom) * 160);
    const float4* wv = (const float4*)(W3e + s * 160);
    float sum = b3e[s];
#pragma unroll 4
    for (int q = 0; q < 20; ++q) {           // 20 x (8 halves vs 2 x float4)
        const uint4  hq = hv[q];
        const float4 w0 = wv[2 * q];
        const float4 w1 = wv[2 * q + 1];
        sum += bf2lo(hq.x) * w0.x + bf2hi(hq.x) * w0.y;
        sum += bf2lo(hq.y) * w0.z + bf2hi(hq.y) * w0.w;
        sum += bf2lo(hq.z) * w1.x + bf2hi(hq.z) * w1.y;
        sum += bf2lo(hq.w) * w1.z + bf2hi(hq.w) * w1.w;
    }
    accum[atom] += 0.125f * sum;
}

// =====================================================================
// Kernel 5: per-batch reduction + SAE (wave32 butterfly)
// =====================================================================
__global__ void finalize_kernel(const float* __restrict__ accum,
                                const int* __restrict__ species,
                                const float* __restrict__ sae,
                                float* __restrict__ out) {
    const int b = blockIdx.x;
    const int a = threadIdx.x;   // 32 threads = 1 wave
    float v = accum[b * ATOMS + a] + sae[species[b * ATOMS + a]];
#pragma unroll
    for (int off = 16; off > 0; off >>= 1)
        v += __shfl_xor(v, off, 32);
    if (a == 0) out[b] = v;
}

// =====================================================================
// Host launcher
// =====================================================================
extern "C" void kernel_launch(void* const* d_in, const int* in_sizes, int n_in,
                              void* d_out, int out_size, void* d_ws, size_t ws_size,
                              hipStream_t stream) {
    (void)in_sizes; (void)n_in; (void)out_size; (void)ws_size;

    const int*   species = (const int*)d_in[0];
    const float* coords  = (const float*)d_in[1];
    const float* W0 = (const float*)d_in[2];
    const float* b0 = (const float*)d_in[3];
    const float* W1 = (const float*)d_in[4];
    const float* b1 = (const float*)d_in[5];
    const float* W2 = (const float*)d_in[6];
    const float* b2 = (const float*)d_in[7];
    const float* W3 = (const float*)d_in[8];
    const float* b3 = (const float*)d_in[9];
    const float* sae = (const float*)d_in[10];
    float* out = (float*)d_out;

    // ---- workspace layout (all 16B-aligned) ----
    char* ws = (char*)d_ws;
    const size_t AEV_B = (size_t)NATOMS * DPAD * 2;              //  4,194,304
    const size_t H1_B  = (size_t)NUM_SPECIES * NATOMS * 256 * 2; //  7,340,032
    const size_t H2_B  = (size_t)NUM_SPECIES * NATOMS * 192 * 2; //  5,505,024
    const size_t H3_B  = (size_t)NUM_SPECIES * NATOMS * 160 * 2; //  4,587,520
    const size_t ACC_B = (size_t)NATOMS * 4;                     //      8,192
    const size_t WT0_B = (size_t)NUM_SPECIES * 32 * 16 * 512 * 2; // 3,670,016
    const size_t WT1_B = (size_t)NUM_SPECIES *  8 * 12 * 512 * 2; //   688,128
    unsigned short* aev = (unsigned short*)ws;
    unsigned short* h1  = (unsigned short*)(ws + AEV_B);
    unsigned short* h2  = (unsigned short*)(ws + AEV_B + H1_B);
    unsigned short* h3  = (unsigned short*)(ws + AEV_B + H1_B + H2_B);
    float*          accum = (float*)(ws + AEV_B + H1_B + H2_B + H3_B);
    unsigned short* wt0 = (unsigned short*)(ws + AEV_B + H1_B + H2_B + H3_B + ACC_B);
    unsigned short* wt1 = (unsigned short*)((char*)wt0 + WT0_B);
    unsigned short* wt2 = (unsigned short*)((char*)wt1 + WT1_B);

    // 1) AEV (also zeroes accum)
    aev_kernel<<<NATOMS, 128, 0, stream>>>(species, coords, aev, accum);

    // 2) MLP ensemble: sequential over e (buffers reused), parallel over
    //    species (grid.y) and tiles (grid.x).
    for (int e = 0; e < 8; ++e) {
        // --- prepack this ensemble's weights into WMMA B-fragments ---
        prepack_kernel<<<(NUM_SPECIES * 32 * 16 * 32 + 255) / 256, 256, 0, stream>>>(
            W0 + (size_t)e * NUM_SPECIES * 1008 * 256, wt0,
            /*Kvalid=*/1008, /*KB=*/32, /*NB=*/16, /*N=*/256);
        prepack_kernel<<<(NUM_SPECIES * 8 * 12 * 32 + 255) / 256, 256, 0, stream>>>(
            W1 + (size_t)e * NUM_SPECIES * 256 * 192, wt1,
            /*Kvalid=*/256, /*KB=*/8, /*NB=*/12, /*N=*/192);
        prepack_kernel<<<(NUM_SPECIES * 6 * 10 * 32 + 255) / 256, 256, 0, stream>>>(
            W2 + (size_t)e * NUM_SPECIES * 192 * 160, wt2,
            /*Kvalid=*/192, /*KB=*/6, /*NB=*/10, /*N=*/160);

        // --- layers ---
        gemm_celu_kernel<<<dim3(16 * 4, NUM_SPECIES), 256, 0, stream>>>(
            aev, DPAD, 0LL, wt0,
            b0 + (size_t)e * NUM_SPECIES * 256,
            h1, 256, /*KB=*/32, /*NB=*/16, /*N=*/256);

        gemm_celu_kernel<<<dim3(16 * 3, NUM_SPECIES), 256, 0, stream>>>(
            h1, 256, (long long)NATOMS * 256, wt1,
            b1 + (size_t)e * NUM_SPECIES * 192,
            h2, 192, /*KB=*/8, /*NB=*/12, /*N=*/192);

        gemm_celu_kernel<<<dim3(16 * 3, NUM_SPECIES), 256, 0, stream>>>(
            h2, 192, (long long)NATOMS * 192, wt2,
            b2 + (size_t)e * NUM_SPECIES * 160,
            h3, 160, /*KB=*/6, /*NB=*/10, /*N=*/160);

        l3_kernel<<<NATOMS / 256, 256, 0, stream>>>(
            h3, species,
            W3 + (size_t)e * NUM_SPECIES * 160,
            b3 + (size_t)e * NUM_SPECIES,
            accum);
    }

    // 3) per-batch energy
    finalize_kernel<<<BATCH, 32, 0, stream>>>(accum, species, sae, out);
}
